// BERT_CRF_266287972432
// MI455X (gfx1250) — matrix operand order repaired and checked by reading, not compiled, not run on hardware
//
#include <hip/hip_runtime.h>
#include <math.h>
#include <stdint.h>

// ---- problem constants ----
#define B_  16
#define S_  256
#define H_  768
#define NH_ 12
#define DH_ 64
#define L_  12
#define FF_ 3072
#define T_  9
#define M_  (B_*S_)   // 4096 tokens

typedef _Float16 half_t;
typedef __attribute__((ext_vector_type(16))) _Float16 v16h;
typedef __attribute__((ext_vector_type(8)))  _Float16 v8h;
typedef __attribute__((ext_vector_type(8)))  float    v8f;

__device__ __forceinline__ v8f wmma16(v16h a, v16h b, v8f c) {
  // D = A(16x32 f16) * B(32x16 f16) + C(16x16 f32)
  return __builtin_amdgcn_wmma_f32_16x16x32_f16(
      /*neg_a=*/false, a, /*neg_b=*/false, b,
      /*c_mod=*/(short)0, c, /*reuse_a=*/false, /*reuse_b=*/false);
}

__device__ __forceinline__ v16h mk16(v8h lo, v8h hi) {
  v16h r;
#pragma unroll
  for (int i = 0; i < 8; i++) { r[i] = lo[i]; r[i + 8] = hi[i]; }
  return r;
}

// ---- CDNA5 async global->LDS copy (ASYNCcnt-tracked), GV addressing mode ----
// LDS dest address for the instruction is the low 32 bits of the generic
// address of the __shared__ location (ISA: LDS aperture truncation).
__device__ __forceinline__ void async_ld_b128(void* lds_dst, const void* gsrc) {
  uint32_t lds_off = (uint32_t)(uintptr_t)lds_dst;
  asm volatile("global_load_async_to_lds_b128 %0, %1, off"
               :: "v"(lds_off), "v"(gsrc)
               : "memory");
}
__device__ __forceinline__ void wait_async0() {
  asm volatile("s_wait_asynccnt 0x0" ::: "memory");
}

__device__ __forceinline__ float gelu_f(float x) {
  return 0.5f * x * (1.0f + erff(x * 0.70710678118654752f));
}

__device__ __forceinline__ float block_sum256(float v, float* red) {
  int t = threadIdx.x;
  red[t] = v;
  __syncthreads();
#pragma unroll
  for (int s = 128; s > 0; s >>= 1) {
    if (t < s) red[t] += red[t + s];
    __syncthreads();
  }
  float r = red[0];
  __syncthreads();
  return r;
}

// ---------------------------------------------------------------------------
// fp32 -> fp16 weight conversion
// ---------------------------------------------------------------------------
__global__ __launch_bounds__(256)
void cvt_f32_f16(const float* __restrict__ src, half_t* __restrict__ dst, int n) {
  int i = (blockIdx.x * 256 + threadIdx.x) * 4;
  if (i + 3 < n) {
    float4 v = *reinterpret_cast<const float4*>(src + i);
    dst[i + 0] = (half_t)v.x; dst[i + 1] = (half_t)v.y;
    dst[i + 2] = (half_t)v.z; dst[i + 3] = (half_t)v.w;
  }
}

// ---------------------------------------------------------------------------
// Embedding gather + LayerNorm  (one 256-thread block per token)
// ---------------------------------------------------------------------------
__global__ __launch_bounds__(256)
void embed_ln(const int* __restrict__ ids, const float* __restrict__ wemb,
              const float* __restrict__ pemb, const float* __restrict__ temb,
              const float* __restrict__ g, const float* __restrict__ bta,
              float* __restrict__ h0, half_t* __restrict__ hh) {
  __shared__ float red[256];
  int tok = blockIdx.x;
  int s   = tok & (S_ - 1);
  int id  = ids[tok];
  float vals[3];
  float sum = 0.0f;
#pragma unroll
  for (int i = 0; i < 3; i++) {
    int hidx = threadIdx.x + i * 256;
    float x = wemb[(size_t)id * H_ + hidx] + pemb[(size_t)s * H_ + hidx] + temb[hidx];
    vals[i] = x; sum += x;
  }
  float mean = block_sum256(sum, red) * (1.0f / H_);
  float sq = 0.0f;
#pragma unroll
  for (int i = 0; i < 3; i++) { float d = vals[i] - mean; sq += d * d; }
  float var  = block_sum256(sq, red) * (1.0f / H_);
  float rstd = rsqrtf(var + 1e-12f);
#pragma unroll
  for (int i = 0; i < 3; i++) {
    int hidx = threadIdx.x + i * 256;
    float y = (vals[i] - mean) * rstd * g[hidx] + bta[hidx];
    h0[(size_t)tok * H_ + hidx] = y;
    hh[(size_t)tok * H_ + hidx] = (half_t)y;
  }
}

// ---------------------------------------------------------------------------
// residual add + LayerNorm : h0 = LN(go + h0); hh = (f16)h0
// ---------------------------------------------------------------------------
__global__ __launch_bounds__(256)
void add_ln(const float* __restrict__ go, const float* __restrict__ g,
            const float* __restrict__ bta, float* __restrict__ h0,
            half_t* __restrict__ hh) {
  __shared__ float red[256];
  int tok = blockIdx.x;
  float vals[3];
  float sum = 0.0f;
#pragma unroll
  for (int i = 0; i < 3; i++) {
    int hidx = threadIdx.x + i * 256;
    float x = go[(size_t)tok * H_ + hidx] + h0[(size_t)tok * H_ + hidx];
    vals[i] = x; sum += x;
  }
  float mean = block_sum256(sum, red) * (1.0f / H_);
  float sq = 0.0f;
#pragma unroll
  for (int i = 0; i < 3; i++) { float d = vals[i] - mean; sq += d * d; }
  float var  = block_sum256(sq, red) * (1.0f / H_);
  float rstd = rsqrtf(var + 1e-12f);
#pragma unroll
  for (int i = 0; i < 3; i++) {
    int hidx = threadIdx.x + i * 256;
    float y = (vals[i] - mean) * rstd * g[hidx] + bta[hidx];
    h0[(size_t)tok * H_ + hidx] = y;
    hh[(size_t)tok * H_ + hidx] = (half_t)y;
  }
}

// ---------------------------------------------------------------------------
// Generic WMMA GEMM:  out = act( A(f16, MxK) @ B(f16, KxN) + bias )
// MODE 0: f16 out; MODE 1: f32 out; MODE 2: gelu -> f16 out
// block tile 128x128, BK=32, 8 waves (wave32), wave tile 32x64.
// A tile staged via global_load_async_to_lds_b128 (ASYNCcnt path);
// B tile transposed through VGPRs into LDS.
// ---------------------------------------------------------------------------
template <int MODE>
__global__ __launch_bounds__(256)
void gemm_wmma(const half_t* __restrict__ A, const half_t* __restrict__ Bw,
               const float* __restrict__ bias, half_t* __restrict__ outH,
               float* __restrict__ outF, int M, int N, int K) {
  constexpr int BK = 32;
  constexpr int AP = BK + 8;            // 40-half pitch = 80B (16B aligned)
  __shared__ half_t Ash[128 * AP];      // [m][k]
  __shared__ half_t Bsh[128 * AP];      // transposed: [n][k]

  const int tid  = threadIdx.x;
  const int lane = tid & 31;
  const int wave = tid >> 5;
  const int wm   = wave & 3;            // 4 waves along M
  const int wn   = wave >> 2;           // 2 waves along N
  const int l15  = lane & 15;
  const int hi   = lane >> 4;
  const int m0   = blockIdx.y * 128;
  const int n0   = blockIdx.x * 128;

  v8f acc[2][4] = {};

  const int ar = tid & 127;             // A-tile row handled by this thread
  const int ac = (tid >> 7) * 16;       // A-tile col chunk
  const int bk = tid & 31;              // B-tile k handled by this thread
  const int bn = (tid >> 5) * 16;       // B-tile n chunk

  for (int k0 = 0; k0 < K; k0 += BK) {
    {   // A tile via async DMA into LDS (two b128 per thread)
      const half_t* src = A + (size_t)(m0 + ar) * K + k0 + ac;
      async_ld_b128(&Ash[ar * AP + ac],     src);
      async_ld_b128(&Ash[ar * AP + ac + 8], src + 8);
    }
    {   // B tile transposed into LDS
      const v8h* src = reinterpret_cast<const v8h*>(Bw + (size_t)(k0 + bk) * N + n0 + bn);
      v8h x0 = src[0], x1 = src[1];
#pragma unroll
      for (int e = 0; e < 8; e++) {
        Bsh[(bn + e) * AP + bk]     = x0[e];
        Bsh[(bn + 8 + e) * AP + bk] = x1[e];
      }
    }
    if (k0 + BK < K) {  // prefetch next tiles while this one is consumed
      __builtin_prefetch(A + (size_t)(m0 + ar) * K + k0 + BK + ac, 0, 1);
      __builtin_prefetch(Bw + (size_t)(k0 + BK + bk) * N + n0 + bn, 0, 1);
    }
    wait_async0();
    __syncthreads();

    v16h af[2], bf[4];
#pragma unroll
    for (int mt = 0; mt < 2; mt++) {
      int row = wm * 32 + mt * 16 + l15;
      v8h c0 = *reinterpret_cast<const v8h*>(&Ash[row * AP + 8 * hi]);
      v8h c1 = *reinterpret_cast<const v8h*>(&Ash[row * AP + 16 + 8 * hi]);
      af[mt] = mk16(c0, c1);
    }
#pragma unroll
    for (int nt = 0; nt < 4; nt++) {
      int col = wn * 64 + nt * 16 + l15;
      v8h c0 = *reinterpret_cast<const v8h*>(&Bsh[col * AP + 8 * hi]);
      v8h c1 = *reinterpret_cast<const v8h*>(&Bsh[col * AP + 16 + 8 * hi]);
      bf[nt] = mk16(c0, c1);
    }
#pragma unroll
    for (int mt = 0; mt < 2; mt++)
#pragma unroll
      for (int nt = 0; nt < 4; nt++)
        acc[mt][nt] = wmma16(af[mt], bf[nt], acc[mt][nt]);
    __syncthreads();
  }

#pragma unroll
  for (int mt = 0; mt < 2; mt++) {
#pragma unroll
    for (int nt = 0; nt < 4; nt++) {
      int gn = n0 + wn * 64 + nt * 16 + l15;
      float bv = bias[gn];
#pragma unroll
      for (int r = 0; r < 8; r++) {
        int gm = m0 + wm * 32 + mt * 16 + r + 8 * hi;
        float v = acc[mt][nt][r] + bv;
        if constexpr (MODE == 2) v = gelu_f(v);
        if constexpr (MODE == 1) outF[(size_t)gm * N + gn] = v;
        else                     outH[(size_t)gm * N + gn] = (half_t)v;
      }
    }
  }
}

// ---------------------------------------------------------------------------
// Fused attention: one workgroup per (batch, head).
// LDS: K (256x64 f16, padded), V^T (64x256 f16, padded), score slabs (8x16x256 f32)
// Dynamic LDS = 201 KB (CDNA5 320KB/WGP budget). K staged via async DMA.
// ---------------------------------------------------------------------------
__global__ __launch_bounds__(256)
void attn_fused(const half_t* __restrict__ q, const half_t* __restrict__ k,
                const half_t* __restrict__ v, const int* __restrict__ am,
                half_t* __restrict__ ctx) {
  constexpr int KP = 72;    // Ksh pitch (halves) : 144B
  constexpr int VP = 264;   // Vt pitch (halves)  : 528B
  constexpr int SP = 264;   // Ssh pitch (floats)
  extern __shared__ char smem[];
  half_t* Ksh = (half_t*)smem;              // 256*KP halves
  half_t* Vt  = Ksh + 256 * KP;             // 64*VP halves
  float*  Ssh = (float*)(Vt + 64 * VP);     // 8*16*SP floats

  const int b    = blockIdx.x / NH_;
  const int head = blockIdx.x % NH_;
  const int tid  = threadIdx.x;
  const int lane = tid & 31;
  const int wave = tid >> 5;
  const int l15  = lane & 15;
  const int hi   = lane >> 4;
  const float scale = 0.125f;   // 1/sqrt(64)

  { // stage K rows (async DMA) and V transposed (VGPR path)
    int s = tid;
    const half_t* krow = k + ((size_t)(b * S_ + s)) * H_ + head * DH_;
    const v8h*    vr   = reinterpret_cast<const v8h*>(v + ((size_t)(b * S_ + s)) * H_ + head * DH_);
#pragma unroll
    for (int c = 0; c < 8; c++)
      async_ld_b128(&Ksh[s * KP + c * 8], krow + c * 8);
#pragma unroll
    for (int c = 0; c < 8; c++) {
      v8h vv = vr[c];
#pragma unroll
      for (int e = 0; e < 8; e++) Vt[(c * 8 + e) * VP + s] = vv[e];
    }
  }
  wait_async0();
  __syncthreads();

  float* Sw = Ssh + wave * 16 * SP;

  for (int pass = 0; pass < 2; pass++) {
    int qr0 = pass * 128 + wave * 16;   // 16 query rows per wave per pass

    // Q A-fragments (16x64 over two K-steps), straight from global
    v16h aq[2];
    {
      const half_t* qbase = q + ((size_t)(b * S_ + qr0 + l15)) * H_ + head * DH_;
#pragma unroll
      for (int ks = 0; ks < 2; ks++) {
        v8h c0 = *reinterpret_cast<const v8h*>(qbase + ks * 32 + 8 * hi);
        v8h c1 = *reinterpret_cast<const v8h*>(qbase + ks * 32 + 16 + 8 * hi);
        aq[ks] = mk16(c0, c1);
      }
    }

    // scores = scale * Q K^T + mask-bias  -> Ssh slab (16x256)
    for (int nt = 0; nt < 16; nt++) {
      v8f acc = {};
      int jrow = nt * 16 + l15;
#pragma unroll
      for (int ks = 0; ks < 2; ks++) {
        v8h c0 = *reinterpret_cast<const v8h*>(&Ksh[jrow * KP + ks * 32 + 8 * hi]);
        v8h c1 = *reinterpret_cast<const v8h*>(&Ksh[jrow * KP + ks * 32 + 16 + 8 * hi]);
        acc = wmma16(aq[ks], mk16(c0, c1), acc);
      }
      float biasv = am[b * S_ + jrow] ? 0.0f : -1.0e9f;
#pragma unroll
      for (int r = 0; r < 8; r++)
        Sw[(r + 8 * hi) * SP + jrow] = acc[r] * scale + biasv;
    }

    // softmax per row: 2 lanes per row, combine via wave32 shuffle
    {
      float* row = Sw + l15 * SP;
      int c0 = hi * 128;
      float mx = -1.0e30f;
      for (int c = 0; c < 128; c++) mx = fmaxf(mx, row[c0 + c]);
      mx = fmaxf(mx, __shfl_xor(mx, 16, 32));
      float sum = 0.0f;
      for (int c = 0; c < 128; c++) {
        float e = expf(row[c0 + c] - mx);
        row[c0 + c] = e;
        sum += e;
      }
      sum += __shfl_xor(sum, 16, 32);
      float inv = 1.0f / sum;
      for (int c = 0; c < 128; c++) row[c0 + c] *= inv;
    }

    // ctx = P @ V : 16 x 64, K = 256 (8 WMMA K-steps)
    v8f co[4] = {};
    for (int ks = 0; ks < 8; ks++) {
      v16h ap;
      const float* prow = Sw + l15 * SP + ks * 32 + 8 * hi;
#pragma unroll
      for (int gidx = 0; gidx < 2; gidx++)
#pragma unroll
        for (int e = 0; e < 8; e++)
          ap[gidx * 8 + e] = (half_t)prow[16 * gidx + e];
#pragma unroll
      for (int nt = 0; nt < 4; nt++) {
        int d = nt * 16 + l15;
        v8h c0 = *reinterpret_cast<const v8h*>(&Vt[d * VP + ks * 32 + 8 * hi]);
        v8h c1 = *reinterpret_cast<const v8h*>(&Vt[d * VP + ks * 32 + 16 + 8 * hi]);
        co[nt] = wmma16(ap, mk16(c0, c1), co[nt]);
      }
    }
#pragma unroll
    for (int nt = 0; nt < 4; nt++) {
      int dg = head * DH_ + nt * 16 + l15;
#pragma unroll
      for (int r = 0; r < 8; r++) {
        int row = qr0 + r + 8 * hi;
        ctx[((size_t)(b * S_ + row)) * H_ + dg] = (half_t)co[nt][r];
      }
    }
  }
}

// ---------------------------------------------------------------------------
// classifier head: logits(M_,T_) = h0 @ cls_W + cls_b   (T_=9, tiny)
// ---------------------------------------------------------------------------
__global__ __launch_bounds__(256)
void cls_head(const float* __restrict__ h0, const float* __restrict__ W,
              const float* __restrict__ bias, float* __restrict__ logits) {
  int i = blockIdx.x * 256 + threadIdx.x;
  if (i >= M_ * T_) return;
  int tok = i / T_;
  int t   = i % T_;
  const float* x = h0 + (size_t)tok * H_;
  float s = bias[t];
  for (int hh = 0; hh < H_; hh++) s += x[hh] * W[hh * T_ + t];
  logits[i] = s;
}

// ---------------------------------------------------------------------------
// CRF negative log-likelihood. One wave32 per batch (16 waves in one block);
// lanes 0..8 carry the 9 tag scores; deterministic ordered reduction.
// ---------------------------------------------------------------------------
__global__ __launch_bounds__(512)
void crf_loss(const float* __restrict__ logits, const int* __restrict__ labels,
              const int* __restrict__ am, const float* __restrict__ start,
              const float* __restrict__ endv, const float* __restrict__ trans,
              float* __restrict__ out) {
  __shared__ float partial[16];
  int lane = threadIdx.x & 31;
  int b    = threadIdx.x >> 5;

  int len = 0;
  for (int s = 0; s < S_; s++) len += am[b * S_ + s];

  const float* em = logits + ((size_t)(b * S_ + 1)) * T_;  // emissions from pos 1
  int j = lane < T_ ? lane : T_ - 1;
  float score = start[j] + em[j];

  for (int t = 1; t < S_ - 1; t++) {
    int pos = t + 1;
    bool mt = (am[b * S_ + pos] != 0) && (pos != len - 1);
    float si[T_];
#pragma unroll
    for (int i = 0; i < T_; i++) si[i] = __shfl(score, i, 32);
    float mx = -1.0e30f;
#pragma unroll
    for (int i = 0; i < T_; i++) mx = fmaxf(mx, si[i] + trans[i * T_ + j]);
    float sum = 0.0f;
#pragma unroll
    for (int i = 0; i < T_; i++) sum += expf(si[i] + trans[i * T_ + j] - mx);
    float nxt = mx + logf(sum) + em[(size_t)t * T_ + j];
    if (mt) score = nxt;
  }

  float mx2 = -1.0e30f;
  float sf[T_];
#pragma unroll
  for (int i = 0; i < T_; i++) {
    sf[i] = __shfl(score, i, 32) + endv[i];
    mx2 = fmaxf(mx2, sf[i]);
  }
  float s2 = 0.0f;
#pragma unroll
  for (int i = 0; i < T_; i++) s2 += expf(sf[i] - mx2);
  float den = mx2 + logf(s2);

  if (lane == 0) {
    int msum = 0, prev = 0;
    float num;
    {
      int pos = 1;
      bool m0 = (am[b * S_ + pos] != 0) && (pos != len - 1);
      int tg0 = m0 ? labels[b * S_ + pos] : 0;
      num = start[tg0] + em[tg0];
      prev = tg0;
      msum += m0 ? 1 : 0;
    }
    for (int t = 1; t < S_ - 1; t++) {
      int pos = t + 1;
      bool mt = (am[b * S_ + pos] != 0) && (pos != len - 1);
      int tg = mt ? labels[b * S_ + pos] : 0;
      if (mt) num += trans[prev * T_ + tg] + em[(size_t)t * T_ + tg];
      prev = tg;
      msum += mt ? 1 : 0;
    }
    int seq_end = msum - 1;
    {
      int pos = 1 + seq_end;
      bool mm = (am[b * S_ + pos] != 0) && (pos != len - 1);
      int lt = mm ? labels[b * S_ + pos] : 0;
      num += endv[lt];
    }
    partial[b] = num - den;
  }
  __syncthreads();
  if (threadIdx.x == 0) {
    float L = 0.0f;
    for (int i = 0; i < 16; i++) L += partial[i];
    out[0] = -L;
  }
}

// ---------------------------------------------------------------------------
// Viterbi decode: one wave32 per batch; lanes 0..8 = tags; backtrack on lane 0.
// ---------------------------------------------------------------------------
__global__ __launch_bounds__(32)
void viterbi_k(const float* __restrict__ logits, const float* __restrict__ start,
               const float* __restrict__ endv, const float* __restrict__ trans,
               float* __restrict__ tags) {
  __shared__ int hist[(S_ - 1) * T_];
  int b    = blockIdx.x;
  int lane = threadIdx.x;
  int j    = lane < T_ ? lane : T_ - 1;
  const float* em = logits + (size_t)b * S_ * T_;
  float score = start[j] + em[j];

  for (int t = 1; t < S_; t++) {
    float si[T_];
#pragma unroll
    for (int i = 0; i < T_; i++) si[i] = __shfl(score, i, 32);
    float best = -1.0e30f; int arg = 0;
#pragma unroll
    for (int i = 0; i < T_; i++) {
      float c = si[i] + trans[i * T_ + j];
      if (c > best) { best = c; arg = i; }
    }
    score = best + em[(size_t)t * T_ + j];
    if (lane < T_) hist[(t - 1) * T_ + j] = arg;
  }

  float bestf = -1.0e30f; int last = 0;
#pragma unroll
  for (int i = 0; i < T_; i++) {
    float c = __shfl(score, i, 32) + endv[i];
    if (c > bestf) { bestf = c; last = i; }
  }
  __syncthreads();
  if (lane == 0) {
    int cur = last;
    tags[(size_t)b * S_ + (S_ - 1)] = (float)cur;
    for (int t = S_ - 2; t >= 0; t--) {
      cur = hist[t * T_ + cur];
      tags[(size_t)b * S_ + t] = (float)cur;
    }
  }
}

// ---------------------------------------------------------------------------
// host-side orchestration
// ---------------------------------------------------------------------------
extern "C" void kernel_launch(void* const* d_in, const int* in_sizes, int n_in,
                              void* d_out, int out_size, void* d_ws, size_t ws_size,
                              hipStream_t stream) {
  (void)in_sizes; (void)n_in; (void)out_size; (void)ws_size;

  const int*   input_ids = (const int*)d_in[0];
  const int*   att_mask  = (const int*)d_in[1];
  const int*   labels    = (const int*)d_in[2];
  const float* word_emb  = (const float*)d_in[3];
  const float* pos_emb   = (const float*)d_in[4];
  const float* type_emb  = (const float*)d_in[5];
  const float* emb_g     = (const float*)d_in[6];
  const float* emb_b     = (const float*)d_in[7];
  const float* lWq = (const float*)d_in[8];
  const float* lbq = (const float*)d_in[9];
  const float* lWk = (const float*)d_in[10];
  const float* lbk = (const float*)d_in[11];
  const float* lWv = (const float*)d_in[12];
  const float* lbv = (const float*)d_in[13];
  const float* lWo = (const float*)d_in[14];
  const float* lbo = (const float*)d_in[15];
  const float* lg1 = (const float*)d_in[16];
  const float* lb1 = (const float*)d_in[17];
  const float* lWi = (const float*)d_in[18];
  const float* lbi = (const float*)d_in[19];
  const float* lWf = (const float*)d_in[20];
  const float* lbf = (const float*)d_in[21];
  const float* lg2 = (const float*)d_in[22];
  const float* lb2 = (const float*)d_in[23];
  const float* cls_W     = (const float*)d_in[24];
  const float* cls_b     = (const float*)d_in[25];
  const float* crf_start = (const float*)d_in[26];
  const float* crf_end   = (const float*)d_in[27];
  const float* crf_trans = (const float*)d_in[28];
  float* out = (float*)d_out;

  // workspace carve-up (needs ~300 MB)
  char* ws = (char*)d_ws;
  size_t off = 0;
  auto alloc = [&](size_t bytes) -> char* {
    char* p = ws + off;
    off = (off + bytes + 255) & ~(size_t)255;
    return p;
  };
  const size_t nHH = (size_t)L_ * H_ * H_;   // 7,077,888
  const size_t nHF = (size_t)L_ * H_ * FF_;  // 28,311,552
  half_t* wq16 = (half_t*)alloc(nHH * 2);
  half_t* wk16 = (half_t*)alloc(nHH * 2);
  half_t* wv16 = (half_t*)alloc(nHH * 2);
  half_t* wo16 = (half_t*)alloc(nHH * 2);
  half_t* wi16 = (half_t*)alloc(nHF * 2);
  half_t* wf16 = (half_t*)alloc(nHF * 2);
  float*  h0   = (float*)alloc((size_t)M_ * H_ * 4);
  half_t* hh   = (half_t*)alloc((size_t)M_ * H_ * 2);
  half_t* qb   = (half_t*)alloc((size_t)M_ * H_ * 2);
  half_t* kb   = (half_t*)alloc((size_t)M_ * H_ * 2);
  half_t* vb   = (half_t*)alloc((size_t)M_ * H_ * 2);
  half_t* ctx  = (half_t*)alloc((size_t)M_ * H_ * 2);
  float*  gf   = (float*)alloc((size_t)M_ * H_ * 4);
  half_t* ff1  = (half_t*)alloc((size_t)M_ * FF_ * 2);
  float*  logits = (float*)alloc((size_t)M_ * T_ * 4);

  // down-convert weights to f16 once per launch (idempotent / deterministic)
  cvt_f32_f16<<<(int)(nHH / 1024), 256, 0, stream>>>(lWq, wq16, (int)nHH);
  cvt_f32_f16<<<(int)(nHH / 1024), 256, 0, stream>>>(lWk, wk16, (int)nHH);
  cvt_f32_f16<<<(int)(nHH / 1024), 256, 0, stream>>>(lWv, wv16, (int)nHH);
  cvt_f32_f16<<<(int)(nHH / 1024), 256, 0, stream>>>(lWo, wo16, (int)nHH);
  cvt_f32_f16<<<(int)(nHF / 1024), 256, 0, stream>>>(lWi, wi16, (int)nHF);
  cvt_f32_f16<<<(int)(nHF / 1024), 256, 0, stream>>>(lWf, wf16, (int)nHF);

  embed_ln<<<M_, 256, 0, stream>>>(input_ids, word_emb, pos_emb, type_emb,
                                   emb_g, emb_b, h0, hh);

  dim3 gHH(H_ / 128, M_ / 128);   // (6, 32)
  dim3 gHF(FF_ / 128, M_ / 128);  // (24, 32)
  const size_t smemAttn = (size_t)256 * 72 * 2 + (size_t)64 * 264 * 2 +
                          (size_t)8 * 16 * 264 * 4;  // 205,824 B

  for (int l = 0; l < L_; l++) {
    const half_t* Wq = wq16 + (size_t)l * H_ * H_;
    const half_t* Wk = wk16 + (size_t)l * H_ * H_;
    const half_t* Wv = wv16 + (size_t)l * H_ * H_;
    const half_t* Wo = wo16 + (size_t)l * H_ * H_;
    const half_t* Wi = wi16 + (size_t)l * H_ * FF_;
    const half_t* Wf = wf16 + (size_t)l * FF_ * H_;

    gemm_wmma<0><<<gHH, 256, 0, stream>>>(hh, Wq, lbq + l * H_, qb, nullptr, M_, H_, H_);
    gemm_wmma<0><<<gHH, 256, 0, stream>>>(hh, Wk, lbk + l * H_, kb, nullptr, M_, H_, H_);
    gemm_wmma<0><<<gHH, 256, 0, stream>>>(hh, Wv, lbv + l * H_, vb, nullptr, M_, H_, H_);
    attn_fused<<<B_ * NH_, 256, smemAttn, stream>>>(qb, kb, vb, att_mask, ctx);
    gemm_wmma<1><<<gHH, 256, 0, stream>>>(ctx, Wo, lbo + l * H_, nullptr, gf, M_, H_, H_);
    add_ln<<<M_, 256, 0, stream>>>(gf, lg1 + l * H_, lb1 + l * H_, h0, hh);
    gemm_wmma<2><<<gHF, 256, 0, stream>>>(hh, Wi, lbi + l * FF_, ff1, nullptr, M_, FF_, H_);
    gemm_wmma<1><<<gHH, 256, 0, stream>>>(ff1, Wf, lbf + l * H_, nullptr, gf, M_, H_, FF_);
    add_ln<<<M_, 256, 0, stream>>>(gf, lg2 + l * H_, lb2 + l * H_, h0, hh);
  }

  cls_head<<<(M_ * T_ + 255) / 256, 256, 0, stream>>>(h0, cls_W, cls_b, logits);
  crf_loss<<<1, 512, 0, stream>>>(logits, labels, att_mask, crf_start, crf_end,
                                  crf_trans, out);
  viterbi_k<<<B_, 32, 0, stream>>>(logits, crf_start, crf_end, crf_trans, out + 1);
}
